// AttnBlock_57415122813391
// MI455X (gfx1250) — compile-verified, matrix operand order checked by reading, not run on hardware
//
#include <hip/hip_runtime.h>
#include <hip/hip_bf16.h>

typedef __attribute__((ext_vector_type(16))) __bf16 v16bf;
typedef __attribute__((ext_vector_type(8)))  __bf16 v8bf_t;
typedef __attribute__((ext_vector_type(8)))  float  v8f;

#define NB    16
#define NC    256
#define NHW   1024
#define NS    77
#define NSP   96
#define NCD   512
#define PR_LD 1064   // self-attn probs LDS row stride (halfs); 532 dwords -> conflict-light
#define PRC_LD 96    // cross-attn probs LDS row stride (halfs)

// Load one 16-half WMMA fragment: lane L -> row/col (L&15), K-runs at
// koff = (L<16?0:8) and koff+16 (ISA 16-bit A/B striping). Two 16B loads.
__device__ __forceinline__ v16bf load_frag(const __bf16* p) {
  union { v16bf v; v8bf_t h[2]; } u;
  u.h[0] = *(const v8bf_t*)(p);
  u.h[1] = *(const v8bf_t*)(p + 16);
  return u.v;
}

__device__ __forceinline__ v8f wmma_bf16f32(v16bf a, v16bf b, v8f c) {
  // D(f32 16x16) = A(bf16 16x32) x B(bf16 32x16) + C
  return __builtin_amdgcn_wmma_f32_16x16x32_bf16(false, a, false, b, (short)0, c,
                                                 false, false);
}

__global__ __launch_bounds__(256) void cvt_bf16_kernel(const float* __restrict__ in,
                                                       __bf16* __restrict__ out, int n) {
  int i = blockIdx.x * 256 + threadIdx.x;
  if (i < n) out[i] = (__bf16)in[i];
}

// GroupNorm(32 groups): one block per (b,g); 8 ch x 1024 pix = 8192 elems.
__global__ __launch_bounds__(256) void groupnorm_kernel(const float* __restrict__ x,
    const float* __restrict__ gamma, const float* __restrict__ beta,
    float* __restrict__ hnf, __bf16* __restrict__ hnb) {
  __shared__ float s1[256], s2[256];
  const int tid = threadIdx.x;
  const int b = blockIdx.x >> 5;
  const int g = blockIdx.x & 31;
  const float* xb = x + (size_t)(b * NC + g * 8) * NHW;
  float s = 0.f, ss = 0.f;
  for (int i = 0; i < 32; i++) {
    float v = xb[i * 256 + tid];
    s += v; ss += v * v;
  }
  s1[tid] = s; s2[tid] = ss;
  __syncthreads();
  for (int st = 128; st > 0; st >>= 1) {
    if (tid < st) { s1[tid] += s1[tid + st]; s2[tid] += s2[tid + st]; }
    __syncthreads();
  }
  const float mean = s1[0] * (1.f / 8192.f);
  const float var  = s2[0] * (1.f / 8192.f) - mean * mean;
  const float rstd = rsqrtf(var + 1e-5f);
  for (int i = 0; i < 32; i++) {
    int f = i * 256 + tid;
    int cc = f >> 10, p = f & 1023;
    int c = g * 8 + cc;
    float v = (xb[f] - mean) * rstd * gamma[c] + beta[c];
    size_t oi = (size_t)(b * NHW + p) * NC + c;
    hnf[oi] = v;
    hnb[oi] = (__bf16)v;
  }
}

// out[m][n] = A[m][:] . W[n][:] + bias[n]   (A: [M,K] bf16, W: [N=256,K] bf16)
// Two-stage software pipeline over K: next-step fragments loaded into a second
// register set before the current step's 8 back-to-back WMMAs are issued, so
// the pre-WMMA wait only covers loads issued a full K-step earlier.
// mode 0: bf16 row-major [m][n]            (q, k, qc, hc)
// mode 1: bf16 transposed [b][n][p]        (v  -> vt)
// mode 2: bf16 [b*96+s][n], pad s>=77 -> 0 (kc)
// mode 3: bf16 [b][n][s] padded            (vc -> vct)
// mode 4: f32 d_out[b][n][p] = x + val     (proj + residual)
__global__ __launch_bounds__(256) void gemm_kernel(const __bf16* __restrict__ A,
    const __bf16* __restrict__ W, const float* __restrict__ bias,
    void* __restrict__ out0, const float* __restrict__ xres, int K, int mode) {
  const int lane = threadIdx.x & 31;
  const int wave = threadIdx.x >> 5;
  const int qsub = wave >> 1;
  const int m0 = blockIdx.x * 64 + qsub * 16;
  const int nh = (wave & 1) * 8;
  const int ln = lane & 15;
  const int koff = (lane < 16) ? 0 : 8;
  const int mrow = m0 + ln;
  long arow = mrow;
  if (mode == 2 || mode == 3) {             // cemb rows: padded 96 -> real 77
    int bb = mrow / NSP, sI = mrow % NSP;
    arow = (long)bb * NS + (sI < NS ? sI : NS - 1);
  }
  const __bf16* ap = A + arow * (long)K + koff;
  const __bf16* wp[8];
#pragma unroll
  for (int t = 0; t < 8; t++)
    wp[t] = W + (long)((nh + t) * 16 + ln) * K + koff;

  v8f acc[8] = {};
  v16bf aC = load_frag(ap);
  v16bf bC[8];
#pragma unroll
  for (int t = 0; t < 8; t++) bC[t] = load_frag(wp[t]);
  for (int ks = 32; ks <= K; ks += 32) {
    const int ksn = (ks < K) ? ks : (K - 32);   // last iter: redundant reload
    v16bf aN = load_frag(ap + ksn);
    v16bf bN[8];
#pragma unroll
    for (int t = 0; t < 8; t++) bN[t] = load_frag(wp[t] + ksn);
#pragma unroll
    for (int t = 0; t < 8; t++) acc[t] = wmma_bf16f32(aC, bC[t], acc[t]);
    aC = aN;
#pragma unroll
    for (int t = 0; t < 8; t++) bC[t] = bN[t];
  }

  for (int t = 0; t < 8; t++) {
    int ncol = (nh + t) * 16 + ln;
    float bia = bias[ncol];
    for (int i = 0; i < 8; i++) {
      int m = m0 + ((lane < 16) ? i : i + 8);  // C-frag: VGPR i -> row i / 8+i
      float val = acc[t][i] + bia;
      if (mode == 0) {
        ((__bf16*)out0)[(size_t)m * NC + ncol] = (__bf16)val;
      } else if (mode == 1) {
        int bb = m >> 10, p = m & 1023;
        ((__bf16*)out0)[(size_t)(bb * NC + ncol) * NHW + p] = (__bf16)val;
      } else if (mode == 2) {
        int sI = m % NSP;
        ((__bf16*)out0)[(size_t)m * NC + ncol] = (__bf16)(sI < NS ? val : 0.f);
      } else if (mode == 3) {
        int bb = m / NSP, sI = m % NSP;
        ((__bf16*)out0)[(size_t)(bb * NC + ncol) * NSP + sI] =
            (__bf16)(sI < NS ? val : 0.f);
      } else {
        int bb = m >> 10, p = m & 1023;
        size_t idx = (size_t)(bb * NC + ncol) * NHW + p;
        ((float*)out0)[idx] = xres[idx] + val;
      }
    }
  }
}

// Fused self-attention: block = (batch, 64-query tile).
// h2 = hn + softmax(Q K^T / 16) V ; probs kept as bf16 in LDS (136 KB).
__global__ __launch_bounds__(256) void self_attn_kernel(const __bf16* __restrict__ q,
    const __bf16* __restrict__ k, const __bf16* __restrict__ vt,
    const float* __restrict__ hnf, float* __restrict__ h2f, __bf16* __restrict__ h2b) {
  __shared__ __bf16 probs[64 * PR_LD];
  __shared__ float red[512];
  const int lane = threadIdx.x & 31;
  const int wave = threadIdx.x >> 5;
  const int ln = lane & 15;
  const int koff = (lane < 16) ? 0 : 8;
  const int bidx = blockIdx.x >> 4;
  const int q0 = (blockIdx.x & 15) * 64;
  const int qsub = wave >> 1;
  const int mrow = q0 + qsub * 16 + ln;
  const float scale = 0.0625f;  // 256^-0.5
  const __bf16* qb = q + (size_t)bidx * NHW * NC;
  const __bf16* kb = k + (size_t)bidx * NHW * NC;
  // Phase 1: S = Q K^T * scale -> LDS (bf16). 64 key-tiles split 8 waves x 4 groups.
  const __bf16* aq = qb + (size_t)mrow * NC + koff;
  for (int g = 0; g < 4; g++) {
    const __bf16* kp[8];
#pragma unroll
    for (int t = 0; t < 8; t++) {
      int key = (g * 16 + (wave & 1) * 8 + t) * 16 + ln;
      kp[t] = kb + (size_t)key * NC + koff;        // B = K^T rows
    }
    v8f acc[8] = {};
    v16bf aC = load_frag(aq);
    v16bf bC[8];
#pragma unroll
    for (int t = 0; t < 8; t++) bC[t] = load_frag(kp[t]);
    for (int ks = 32; ks <= NC; ks += 32) {
      const int ksn = (ks < NC) ? ks : (NC - 32);
      v16bf aN = load_frag(aq + ksn);
      v16bf bN[8];
#pragma unroll
      for (int t = 0; t < 8; t++) bN[t] = load_frag(kp[t] + ksn);
#pragma unroll
      for (int t = 0; t < 8; t++) acc[t] = wmma_bf16f32(aC, bC[t], acc[t]);
      aC = aN;
#pragma unroll
      for (int t = 0; t < 8; t++) bC[t] = bN[t];
    }
    for (int t = 0; t < 8; t++) {
      int col = (g * 16 + (wave & 1) * 8 + t) * 16 + ln;
      for (int i = 0; i < 8; i++) {
        int row = qsub * 16 + ((lane < 16) ? i : i + 8);
        probs[row * PR_LD + col] = (__bf16)(acc[t][i] * scale);
      }
    }
  }
  __syncthreads();
  // Phase 2: softmax, 64 rows x 4 threads/row (256-elem segments).
  {
    const int row = threadIdx.x >> 2;
    const int part = threadIdx.x & 3;
    __bf16* pr = &probs[row * PR_LD + part * 256];
    float mx = -3.0e38f;
    for (int j = 0; j < 256; j++) mx = fmaxf(mx, (float)pr[j]);
    red[row * 4 + part] = mx;
    __syncthreads();
    float rmax = fmaxf(fmaxf(red[row * 4], red[row * 4 + 1]),
                       fmaxf(red[row * 4 + 2], red[row * 4 + 3]));
    float sum = 0.f;
    for (int j = 0; j < 256; j++) {
      float e = __expf((float)pr[j] - rmax);
      pr[j] = (__bf16)e;
      sum += e;
    }
    red[256 + row * 4 + part] = sum;
    __syncthreads();
    float rsum = red[256 + row * 4] + red[256 + row * 4 + 1] +
                 red[256 + row * 4 + 2] + red[256 + row * 4 + 3];
    float inv = 1.f / rsum;
    for (int j = 0; j < 256; j++) pr[j] = (__bf16)((float)pr[j] * inv);
  }
  __syncthreads();
  // Phase 3: O = P V (K = 1024 keys); B operand from vt[b][ch][key] (contiguous).
  const __bf16* vtb = vt + (size_t)bidx * NC * NHW;
  const __bf16* ap = &probs[(qsub * 16 + ln) * PR_LD + koff];
  const __bf16* vp[8];
#pragma unroll
  for (int t = 0; t < 8; t++) {
    int ch = ((wave & 1) * 8 + t) * 16 + ln;
    vp[t] = vtb + (size_t)ch * NHW + koff;
  }
  v8f acc[8] = {};
  v16bf aC = load_frag(ap);
  v16bf bC[8];
#pragma unroll
  for (int t = 0; t < 8; t++) bC[t] = load_frag(vp[t]);
  for (int ks = 32; ks <= NHW; ks += 32) {
    const int ksn = (ks < NHW) ? ks : (NHW - 32);
    v16bf aN = load_frag(ap + ksn);
    v16bf bN[8];
#pragma unroll
    for (int t = 0; t < 8; t++) bN[t] = load_frag(vp[t] + ksn);
#pragma unroll
    for (int t = 0; t < 8; t++) acc[t] = wmma_bf16f32(aC, bC[t], acc[t]);
    aC = aN;
#pragma unroll
    for (int t = 0; t < 8; t++) bC[t] = bN[t];
  }
  for (int t = 0; t < 8; t++) {
    int ch = ((wave & 1) * 8 + t) * 16 + ln;
    for (int i = 0; i < 8; i++) {
      int row = q0 + qsub * 16 + ((lane < 16) ? i : i + 8);
      size_t idx = (size_t)(bidx * NHW + row) * NC + ch;
      float val = hnf[idx] + acc[t][i];   // residual onto normed h
      h2f[idx] = val;
      h2b[idx] = (__bf16)val;
    }
  }
}

// Fused cross-attention: keys padded 77 -> 96 (masked in softmax, zeroed probs).
__global__ __launch_bounds__(256) void cross_attn_kernel(const __bf16* __restrict__ qc,
    const __bf16* __restrict__ kc, const __bf16* __restrict__ vct,
    __bf16* __restrict__ hc) {
  __shared__ __bf16 probs[64 * PRC_LD];
  const int lane = threadIdx.x & 31;
  const int wave = threadIdx.x >> 5;
  const int ln = lane & 15;
  const int koff = (lane < 16) ? 0 : 8;
  const int bidx = blockIdx.x >> 4;
  const int q0 = (blockIdx.x & 15) * 64;
  const int qsub = wave >> 1;
  const int mrow = q0 + qsub * 16 + ln;
  const float scale = 0.0625f;
  const __bf16* qcb = qc + (size_t)bidx * NHW * NC;
  const __bf16* kcb = kc + (size_t)bidx * NSP * NC;
  {
    v8f acc[3] = {};   // 6 key tiles split across the 2 waves of each qsub
    for (int ks = 0; ks < NC; ks += 32) {
      v16bf a = load_frag(qcb + (size_t)mrow * NC + ks + koff);
      v16bf bfr[3];
      for (int t = 0; t < 3; t++) {
        int key = ((wave & 1) * 3 + t) * 16 + ln;
        bfr[t] = load_frag(kcb + (size_t)key * NC + ks + koff);
      }
      for (int t = 0; t < 3; t++)
        acc[t] = wmma_bf16f32(a, bfr[t], acc[t]);
    }
    for (int t = 0; t < 3; t++) {
      int col = ((wave & 1) * 3 + t) * 16 + ln;
      for (int i = 0; i < 8; i++) {
        int row = qsub * 16 + ((lane < 16) ? i : i + 8);
        probs[row * PRC_LD + col] = (__bf16)(acc[t][i] * scale);
      }
    }
  }
  __syncthreads();
  if (threadIdx.x < 64) {   // one thread per row; only first 77 keys are real
    __bf16* pr = &probs[threadIdx.x * PRC_LD];
    float mx = -3.0e38f;
    for (int j = 0; j < NS; j++) mx = fmaxf(mx, (float)pr[j]);
    float sum = 0.f;
    for (int j = 0; j < NS; j++) {
      float e = __expf((float)pr[j] - mx);
      pr[j] = (__bf16)e;
      sum += e;
    }
    float inv = 1.f / sum;
    for (int j = 0; j < NS; j++) pr[j] = (__bf16)((float)pr[j] * inv);
    for (int j = NS; j < NSP; j++) pr[j] = (__bf16)0.f;
  }
  __syncthreads();
  const __bf16* vcb = vct + (size_t)bidx * NC * NSP;
  v8f acc[8] = {};
  for (int ks = 0; ks < NSP; ks += 32) {   // K = 96: 3 steps
    v16bf a = load_frag(&probs[(qsub * 16 + ln) * PRC_LD + ks + koff]);
    v16bf bfr[8];
    for (int t = 0; t < 8; t++) {
      int ch = ((wave & 1) * 8 + t) * 16 + ln;
      bfr[t] = load_frag(vcb + (size_t)ch * NSP + ks + koff);
    }
    for (int t = 0; t < 8; t++)
      acc[t] = wmma_bf16f32(a, bfr[t], acc[t]);
  }
  for (int t = 0; t < 8; t++) {
    int ch = ((wave & 1) * 8 + t) * 16 + ln;
    for (int i = 0; i < 8; i++) {
      int row = q0 + qsub * 16 + ((lane < 16) ? i : i + 8);
      hc[(size_t)(bidx * NHW + row) * NC + ch] = (__bf16)acc[t][i];
    }
  }
}

extern "C" void kernel_launch(void* const* d_in, const int* in_sizes, int n_in,
                              void* d_out, int out_size, void* d_ws, size_t ws_size,
                              hipStream_t stream) {
  (void)in_sizes; (void)n_in; (void)out_size; (void)ws_size;
  const float* x      = (const float*)d_in[0];
  const float* cemb   = (const float*)d_in[1];
  const float* gamma  = (const float*)d_in[2];
  const float* beta   = (const float*)d_in[3];
  const float* wq_s   = (const float*)d_in[4];
  const float* bq_s   = (const float*)d_in[5];
  const float* wk_s   = (const float*)d_in[6];
  const float* bk_s   = (const float*)d_in[7];
  const float* wv_s   = (const float*)d_in[8];
  const float* bv_s   = (const float*)d_in[9];
  const float* wq_c   = (const float*)d_in[10];
  const float* bq_c   = (const float*)d_in[11];
  const float* wk_c   = (const float*)d_in[12];
  const float* bk_c   = (const float*)d_in[13];
  const float* wv_c   = (const float*)d_in[14];
  const float* bv_c   = (const float*)d_in[15];
  const float* w_proj = (const float*)d_in[16];
  const float* b_proj = (const float*)d_in[17];
  float* out = (float*)d_out;

  char* base = (char*)d_ws;
  size_t off = 0;
  auto alloc = [&](size_t bytes) -> void* {
    void* p = base + off;
    off += (bytes + 255) & ~(size_t)255;
    return p;
  };
  const size_t TOK = (size_t)NB * NHW;  // 16384 tokens
  float*  hnf   = (float*) alloc(TOK * NC * 4);
  __bf16* hnb   = (__bf16*)alloc(TOK * NC * 2);
  __bf16* qb    = (__bf16*)alloc(TOK * NC * 2);
  __bf16* kb    = (__bf16*)alloc(TOK * NC * 2);
  __bf16* vtb   = (__bf16*)alloc(TOK * NC * 2);   // V transposed [b][c][p]
  float*  h2f   = (float*) alloc(TOK * NC * 4);
  __bf16* h2b   = (__bf16*)alloc(TOK * NC * 2);
  __bf16* qcb   = (__bf16*)alloc(TOK * NC * 2);
  __bf16* kcb   = (__bf16*)alloc((size_t)NB * NSP * NC * 2);
  __bf16* vctb  = (__bf16*)alloc((size_t)NB * NC * NSP * 2); // Vc^T padded
  __bf16* hcb   = (__bf16*)alloc(TOK * NC * 2);
  __bf16* cembb = (__bf16*)alloc((size_t)NB * NS * NCD * 2);
  __bf16* wqsb  = (__bf16*)alloc((size_t)NC * NC * 2);
  __bf16* wksb  = (__bf16*)alloc((size_t)NC * NC * 2);
  __bf16* wvsb  = (__bf16*)alloc((size_t)NC * NC * 2);
  __bf16* wqcb  = (__bf16*)alloc((size_t)NC * NC * 2);
  __bf16* wkcb  = (__bf16*)alloc((size_t)NC * NCD * 2);
  __bf16* wvcb  = (__bf16*)alloc((size_t)NC * NCD * 2);
  __bf16* wpb   = (__bf16*)alloc((size_t)NC * NC * 2);

  auto cvt = [&](const float* in, __bf16* o, int n) {
    cvt_bf16_kernel<<<(n + 255) / 256, 256, 0, stream>>>(in, o, n);
  };
  cvt(cemb, cembb, NB * NS * NCD);
  cvt(wq_s, wqsb, NC * NC);
  cvt(wk_s, wksb, NC * NC);
  cvt(wv_s, wvsb, NC * NC);
  cvt(wq_c, wqcb, NC * NC);
  cvt(wk_c, wkcb, NC * NCD);
  cvt(wv_c, wvcb, NC * NCD);
  cvt(w_proj, wpb, NC * NC);

  groupnorm_kernel<<<NB * 32, 256, 0, stream>>>(x, gamma, beta, hnf, hnb);

  gemm_kernel<<<TOK / 64, 256, 0, stream>>>(hnb, wqsb, bq_s, qb,  nullptr, NC, 0);
  gemm_kernel<<<TOK / 64, 256, 0, stream>>>(hnb, wksb, bk_s, kb,  nullptr, NC, 0);
  gemm_kernel<<<TOK / 64, 256, 0, stream>>>(hnb, wvsb, bv_s, vtb, nullptr, NC, 1);

  self_attn_kernel<<<NB * 16, 256, 0, stream>>>(qb, kb, vtb, hnf, h2f, h2b);

  gemm_kernel<<<TOK / 64, 256, 0, stream>>>(h2b, wqcb, bq_c, qcb, nullptr, NC, 0);
  gemm_kernel<<<(NB * NSP) / 64, 256, 0, stream>>>(cembb, wkcb, bk_c, kcb,  nullptr, NCD, 2);
  gemm_kernel<<<(NB * NSP) / 64, 256, 0, stream>>>(cembb, wvcb, bv_c, vctb, nullptr, NCD, 3);

  cross_attn_kernel<<<NB * 16, 256, 0, stream>>>(qcb, kcb, vctb, hcb);

  gemm_kernel<<<TOK / 64, 256, 0, stream>>>(hcb, wpb, b_proj, out, x, NC, 4);
}
